// GRUPredictor_80985903333837
// MI455X (gfx1250) — matrix operand order repaired and checked by reading, not compiled, not run on hardware
//
#include <hip/hip_runtime.h>

// ---------------------------------------------------------------------------
// GRU encoder-decoder for MI455X (gfx1250), f16 WMMA GEMMs + fused GRU gates.
// B=1024, T=168, IN=OUT=64, H=1024, L=2, TGT=24.
// Hidden state kept in f32 (recurrence) + f16 (WMMA A operand) so the hot
// K=1024 GEMM phases run conversion-free straight from L2-resident f16.
// ---------------------------------------------------------------------------

typedef __attribute__((ext_vector_type(16))) _Float16 v16h;
typedef __attribute__((ext_vector_type(8)))  _Float16 v8h;
typedef __attribute__((ext_vector_type(4)))  _Float16 v4h;
typedef __attribute__((ext_vector_type(8)))  float    v8f;

#define H_DIM   1024
#define B_DIM   1024
#define T_DIM   168
#define TGT_DIM 24
#define IN_DIM  64
#define OUT_DIM 64

// Assemble a 16-half WMMA fragment from two 16-byte loads (VGPRs 0..3, 4..7).
static __device__ __forceinline__ v16h make_frag(v8h lo, v8h hi) {
  v16h f;
#pragma unroll
  for (int i = 0; i < 8; ++i) { f[i] = lo[i]; f[i + 8] = hi[i]; }
  return f;
}

static __device__ __forceinline__ float fast_sigmoid(float x) {
  return 1.0f / (1.0f + __expf(-x));
}
static __device__ __forceinline__ float fast_tanh(float x) {
  return 1.0f - 2.0f / (__expf(2.0f * x) + 1.0f);
}

// One GEMM phase: accumulate A[128 x K] @ W[g*H + cols, K]^T into r/z/n accums.
//  A_IS_F32 = true : A is f32 in global, staged through LDS as f16 (layer-0
//                    input phase only, K = 64).
//  A_IS_F32 = false: A is f16 row-major in global; fragments loaded directly
//                    with two global b128 loads per 16x32 tile (hot path).
template <bool A_IS_F32>
static __device__ __forceinline__ void gemm_phase(
    const float* __restrict__ Af, const _Float16* __restrict__ Ah, long lda,
    int K, const _Float16* __restrict__ W,
    _Float16* sh,                 // LDS staging buffer, 128*32 halves
    int rowbase_block,            // blockIdx.y * 128
    int colbase,                  // blockIdx.x * 64 + wave_n * 32 (per-gate col)
    int wave_m, int tid, int lane,
    v8f (&acc_r)[2][2], v8f (&acc_z)[2][2], v8f (&acc_n)[2][2]) {
  const int half  = lane >> 4;          // 0: lanes 0-15, 1: lanes 16-31
  const int nloc  = lane & 15;
  const int arow0 = wave_m * 32 + nloc; // local row of A fragment 0

  // Hoist the 6 weight-tile base pointers (3 gates x 2 col-tiles) out of the
  // K loop; inner loop addressing is then pure pointer + k0.
  const _Float16* bp_base[3][2];
#pragma unroll
  for (int g = 0; g < 3; ++g)
#pragma unroll
    for (int ni = 0; ni < 2; ++ni) {
      const int wrow = g * H_DIM + colbase + ni * 16 + nloc;
      bp_base[g][ni] = W + (long)wrow * K + 8 * half;
    }

  for (int k0 = 0; k0 < K; k0 += 32) {
    v16h afrag[2];
    if (A_IS_F32) {
      // ---- stage A chunk [128 x 32] f32 -> f16 into LDS (vectorized) ----
      const int r0 = tid >> 3;          // each thread: 4 rows, 4 cols each
      const int c0 = (tid & 7) * 4;
#pragma unroll
      for (int i = 0; i < 4; ++i) {
        const int r = r0 + i * 32;
        const float4 v =
            *(const float4*)(Af + (long)(rowbase_block + r) * lda + k0 + c0);
        v4h h4;
        h4[0] = (_Float16)v.x; h4[1] = (_Float16)v.y;
        h4[2] = (_Float16)v.z; h4[3] = (_Float16)v.w;
        *(v4h*)(sh + r * 32 + c0) = h4;
      }
      __syncthreads();
      // A layout: half 0 -> K 0-7,16-23; half 1 -> K 8-15,24-31
#pragma unroll
      for (int mi = 0; mi < 2; ++mi) {
        const _Float16* ap = sh + (arow0 + mi * 16) * 32 + 8 * half;
        afrag[mi] = make_frag(*(const v8h*)ap, *(const v8h*)(ap + 16));
      }
    } else {
      // ---- direct f16 A fragments from global (no LDS, no cvt) ----
#pragma unroll
      for (int mi = 0; mi < 2; ++mi) {
        const _Float16* ap =
            Ah + (long)(rowbase_block + arow0 + mi * 16) * lda + k0 + 8 * half;
        afrag[mi] = make_frag(*(const v8h*)ap, *(const v8h*)(ap + 16));
      }
    }

    // ---- 3 gates x 2 col-tiles x 2 row-tiles of WMMA ----
#pragma unroll
    for (int g = 0; g < 3; ++g) {
      v8f (*accsel)[2] = (g == 0) ? acc_r : (g == 1) ? acc_z : acc_n;
#pragma unroll
      for (int ni = 0; ni < 2; ++ni) {
        const _Float16* bp = bp_base[g][ni] + k0;
        // WGP-scope prefetch of the next K chunk: lands in WGP$ where the
        // 4 waves sharing this B tile re-hit it (locality=3 -> near policy).
        if (k0 + 32 < K) __builtin_prefetch(bp + 32, 0, 3);
        v16h bfrag = make_frag(*(const v8h*)bp, *(const v8h*)(bp + 16));
#pragma unroll
        for (int mi = 0; mi < 2; ++mi) {
          accsel[mi][ni] = __builtin_amdgcn_wmma_f32_16x16x32_f16(
              false, afrag[mi], false, bfrag, (short)0, accsel[mi][ni],
              false, false);
        }
      }
    }
    if (A_IS_F32) __syncthreads();
  }
}

// Fused GRU layer step: h_out = GRU(inp, h_prev) for one layer, one time step.
// Grid: (16 N-blocks of 64 cols, 8 M-blocks of 128 rows), block = 256 threads.
template <bool A_IS_F32>
__global__ __launch_bounds__(256) void gru_layer_kernel(
    const float* __restrict__ inpf,      // used when A_IS_F32
    const _Float16* __restrict__ inph,   // used when !A_IS_F32
    long lda, int Kin,
    const _Float16* __restrict__ Wih,    // [3H x Kin] f16 row-major
    const float* __restrict__ hprev,     // [B x H] f32 (recurrence)
    const _Float16* __restrict__ hprev16,// [B x H] f16 (WMMA A operand)
    const _Float16* __restrict__ Whh,    // [3H x H]  f16 row-major
    const float* __restrict__ bih, const float* __restrict__ bhh,
    float* __restrict__ hout, _Float16* __restrict__ hout16) {
  __shared__ _Float16 sh[128 * 32];

  const int tid    = threadIdx.x;
  const int lane   = tid & 31;
  const int wave   = tid >> 5;
  const int wave_m = wave >> 1;   // 0..3 -> 32-row strip
  const int wave_n = wave & 1;    // 0..1 -> 32-col strip
  const int rowbase_block = blockIdx.y * 128;
  const int colbase       = blockIdx.x * 64 + wave_n * 32;

  v8f acc_r[2][2]  = {};
  v8f acc_z[2][2]  = {};
  v8f acc_nx[2][2] = {};
  v8f acc_nh[2][2] = {};

  // gx = inp @ Wih^T   (n-gate part into acc_nx)
  gemm_phase<A_IS_F32>(inpf, inph, lda, Kin, Wih, sh, rowbase_block, colbase,
                       wave_m, tid, lane, acc_r, acc_z, acc_nx);
  // gh = hprev @ Whh^T (n-gate part into acc_nh) — always direct f16
  gemm_phase<false>(nullptr, hprev16, (long)H_DIM, H_DIM, Whh, sh,
                    rowbase_block, colbase, wave_m, tid, lane, acc_r, acc_z,
                    acc_nh);

  // Fused GRU gates.  C layout: VGPR e <-> M = e + 8*(lane/16), N = lane%16.
  const int half = lane >> 4;
  const int nloc = lane & 15;
#pragma unroll
  for (int ni = 0; ni < 2; ++ni) {
    const int col   = colbase + ni * 16 + nloc;
    const float br  = bih[col] + bhh[col];
    const float bz  = bih[H_DIM + col] + bhh[H_DIM + col];
    const float bnx = bih[2 * H_DIM + col];
    const float bnh = bhh[2 * H_DIM + col];
#pragma unroll
    for (int mi = 0; mi < 2; ++mi) {
#pragma unroll
      for (int e = 0; e < 8; ++e) {
        const int b    = rowbase_block + wave_m * 32 + mi * 16 + e + 8 * half;
        const float hv = hprev[(long)b * H_DIM + col];
        const float rv = fast_sigmoid(acc_r[mi][ni][e] + br);
        const float zv = fast_sigmoid(acc_z[mi][ni][e] + bz);
        const float nv =
            fast_tanh(acc_nx[mi][ni][e] + bnx + rv * (acc_nh[mi][ni][e] + bnh));
        const float hnew = (1.0f - zv) * nv + zv * hv;
        hout[(long)b * H_DIM + col]   = hnew;
        hout16[(long)b * H_DIM + col] = (_Float16)hnew;
      }
    }
  }
}

// Output head: pred[b,o] = top[b,:] . Wfc[o,:] + bfc[o]; writes d_out and the
// f32 decoder feedback buffer.  3 GFLOP total across 24 steps -> plain f32.
__global__ __launch_bounds__(256) void head_kernel(
    const float* __restrict__ top, const float* __restrict__ Wfc,
    const float* __restrict__ bfc, float* __restrict__ out,
    float* __restrict__ cur, int t) {
  const int idx = blockIdx.x * 256 + threadIdx.x;   // 0 .. B*OUT-1
  const int b = idx >> 6;
  const int o = idx & 63;
  const float* tr = top + (long)b * H_DIM;
  const float* wr = Wfc + (long)o * H_DIM;
  float s = bfc[o];
  for (int k = 0; k < H_DIM; k += 4) {
    const float4 tv = *(const float4*)(tr + k);
    const float4 wv = *(const float4*)(wr + k);
    s += tv.x * wv.x + tv.y * wv.y + tv.z * wv.z + tv.w * wv.w;
  }
  out[((long)b * TGT_DIM + t) * OUT_DIM + o] = s;
  cur[(long)b * OUT_DIM + o] = s;
}

__global__ void cvt_f32_to_f16(const float* __restrict__ src,
                               _Float16* __restrict__ dst, int n) {
  int i = blockIdx.x * 256 + threadIdx.x;
  if (i < n) dst[i] = (_Float16)src[i];
}

__global__ void zero_f32(float* __restrict__ p, int n) {
  int i = blockIdx.x * 256 + threadIdx.x;
  if (i < n) p[i] = 0.0f;
}

extern "C" void kernel_launch(void* const* d_in, const int* in_sizes, int n_in,
                              void* d_out, int out_size, void* d_ws,
                              size_t ws_size, hipStream_t stream) {
  const float* x    = (const float*)d_in[0];
  const float* Wih0 = (const float*)d_in[1];
  const float* Whh0 = (const float*)d_in[2];
  const float* bih0 = (const float*)d_in[3];
  const float* bhh0 = (const float*)d_in[4];
  const float* Wih1 = (const float*)d_in[5];
  const float* Whh1 = (const float*)d_in[6];
  const float* bih1 = (const float*)d_in[7];
  const float* bhh1 = (const float*)d_in[8];
  const float* Wfc  = (const float*)d_in[9];
  const float* bfc  = (const float*)d_in[10];
  float* out = (float*)d_out;

  // Workspace layout (~45 MB): f16 weights, f32 hidden (double buffered),
  // f16 hidden (double buffered), f32 decoder feedback.
  const size_t nWih0 = (size_t)3 * H_DIM * IN_DIM;
  const size_t nWhh  = (size_t)3 * H_DIM * H_DIM;
  const size_t nH    = (size_t)B_DIM * H_DIM;
  _Float16* Wih0h = (_Float16*)d_ws;
  _Float16* Whh0h = Wih0h + nWih0;
  _Float16* Wih1h = Whh0h + nWhh;
  _Float16* Whh1h = Wih1h + nWhh;
  float* h0a = (float*)(Whh1h + nWhh);
  float* h0b = h0a + nH;
  float* h1a = h0b + nH;
  float* h1b = h1a + nH;
  _Float16* h0a16 = (_Float16*)(h1b + nH);
  _Float16* h0b16 = h0a16 + nH;
  _Float16* h1a16 = h0b16 + nH;
  _Float16* h1b16 = h1a16 + nH;
  float* cur = (float*)(h1b16 + nH);

  dim3 blk(256);

  // Weights -> f16 (L2-resident for the whole sequence).
  cvt_f32_to_f16<<<(int)((nWih0 + 255) / 256), blk, 0, stream>>>(Wih0, Wih0h, (int)nWih0);
  cvt_f32_to_f16<<<(int)((nWhh  + 255) / 256), blk, 0, stream>>>(Whh0, Whh0h, (int)nWhh);
  cvt_f32_to_f16<<<(int)((nWhh  + 255) / 256), blk, 0, stream>>>(Wih1, Wih1h, (int)nWhh);
  cvt_f32_to_f16<<<(int)((nWhh  + 255) / 256), blk, 0, stream>>>(Whh1, Whh1h, (int)nWhh);

  // h starts at zero: 4 f32 buffers + 4 f16 buffers (f16 region = 2M floats).
  zero_f32<<<(int)((4 * nH + 255) / 256), blk, 0, stream>>>(h0a, (int)(4 * nH));
  zero_f32<<<(int)((2 * nH + 255) / 256), blk, 0, stream>>>((float*)h0a16,
                                                            (int)(2 * nH));

  dim3 ggrid(16, 8);   // 16 N-blocks x 8 M-blocks = 128 workgroups
  float *h0 = h0a, *h0n = h0b, *h1 = h1a, *h1n = h1b;
  _Float16 *h0_16 = h0a16, *h0n_16 = h0b16, *h1_16 = h1a16, *h1n_16 = h1b16;

  // ---- Encoder: 168 sequential steps of the 2-layer stack ----
  for (int t = 0; t < T_DIM; ++t) {
    gru_layer_kernel<true><<<ggrid, blk, 0, stream>>>(
        x + (long)t * IN_DIM, nullptr, (long)T_DIM * IN_DIM, IN_DIM,
        Wih0h, h0, h0_16, Whh0h, bih0, bhh0, h0n, h0n_16);
    gru_layer_kernel<false><<<ggrid, blk, 0, stream>>>(
        nullptr, h0n_16, (long)H_DIM, H_DIM,
        Wih1h, h1, h1_16, Whh1h, bih1, bhh1, h1n, h1n_16);
    float* t1; _Float16* t2;
    t1 = h0; h0 = h0n; h0n = t1;   t2 = h0_16; h0_16 = h0n_16; h0n_16 = t2;
    t1 = h1; h1 = h1n; h1n = t1;   t2 = h1_16; h1_16 = h1n_16; h1n_16 = t2;
  }

  // ---- Autoregressive decoder: 24 steps, prediction fed back ----
  for (int t = 0; t < TGT_DIM; ++t) {
    const float* inp = (t == 0) ? (x + (long)(T_DIM - 1) * IN_DIM) : cur;
    const long   lda = (t == 0) ? (long)T_DIM * IN_DIM : (long)IN_DIM;
    gru_layer_kernel<true><<<ggrid, blk, 0, stream>>>(
        inp, nullptr, lda, IN_DIM,
        Wih0h, h0, h0_16, Whh0h, bih0, bhh0, h0n, h0n_16);
    gru_layer_kernel<false><<<ggrid, blk, 0, stream>>>(
        nullptr, h0n_16, (long)H_DIM, H_DIM,
        Wih1h, h1, h1_16, Whh1h, bih1, bhh1, h1n, h1n_16);
    head_kernel<<<(B_DIM * OUT_DIM) / 256, blk, 0, stream>>>(
        h1n, Wfc, bfc, out, cur, t);
    float* t1; _Float16* t2;
    t1 = h0; h0 = h0n; h0n = t1;   t2 = h0_16; h0_16 = h0n_16; h0n_16 = t2;
    t1 = h1; h1 = h1n; h1n = t1;   t2 = h1_16; h1_16 = h1n_16; h1n_16 = t2;
  }
}
